// MsaHmmLayer_25168508354911
// MI455X (gfx1250) — matrix operand order repaired and checked by reading, not compiled, not run on hardware
//
#include <hip/hip_runtime.h>
#include <math.h>

#define MDIM   4
#define BATCH  32
#define TSTEPS 512
#define SDIM   26
#define SPAD   32
#define QDIM   512
#define BROWS  16   // batch rows handled per block (2 blocks per model m)

typedef __attribute__((ext_vector_type(16))) __bf16 v16bf;
typedef __attribute__((ext_vector_type(8)))  float  v8f;
typedef __attribute__((ext_vector_type(4)))  int    i4x;

union FragB { v16bf v; i4x d[2]; };

__device__ __forceinline__ unsigned short f2bf(float f) {
  unsigned u = __float_as_uint(f);
  u += 0x7fffu + ((u >> 16) & 1u);          // round to nearest even
  return (unsigned short)(u >> 16);
}

// WGP-scope prefetch (no scope modifier -> SCOPE 0 = WGP): fills all cache
// levels. __builtin_prefetch only reaches SE/SYS scope (stops at GL2).
// Used ONLY off the critical path (x row, a full timestep of lead time).
__device__ __forceinline__ void prefetch_wgp(const void* p) {
  asm volatile("global_prefetch_b8 %0, off"
               :
               : "v"((unsigned long long)(uintptr_t)p));
}

// One-time prep: At[m][p][q] = bf16(A[m][q][p])  (transposed transition matrix)
//                Bp[m][q][s] = bf16(B[m][q][s]) padded S:26 -> 32 with zeros
__global__ void __launch_bounds__(256)
hmm_convert_kernel(const float* __restrict__ A, const float* __restrict__ B,
                   unsigned short* __restrict__ At, unsigned short* __restrict__ Bp) {
  int idx = blockIdx.x * blockDim.x + threadIdx.x;
  const int totalA = MDIM * QDIM * QDIM;
  if (idx < totalA) {
    int m = idx >> 18;
    int r = idx & (QDIM * QDIM - 1);
    int p = r >> 9;
    int q = r & (QDIM - 1);
    At[idx] = f2bf(A[((size_t)m << 18) + ((size_t)q << 9) + p]);
  } else {
    int j = idx - totalA;
    if (j < MDIM * QDIM * SPAD) {
      int m = j >> 14;
      int r = j & (QDIM * SPAD - 1);
      int q = r >> 5;
      int s = r & 31;
      Bp[j] = (s < SDIM) ? f2bf(B[((size_t)m * QDIM + q) * SDIM + s])
                         : (unsigned short)0;
    }
  }
}

// Forward recursion. grid = 2*MDIM blocks; block (m, batch-half) keeps its
// 16x512 alpha slice in LDS (bf16) across all 512 serial steps.
__global__ void __launch_bounds__(512)
hmm_forward_kernel(const float* __restrict__ x, const float* __restrict__ initp,
                   const unsigned short* __restrict__ At,
                   const unsigned short* __restrict__ Bp,
                   float* __restrict__ out) {
  extern __shared__ char smem_raw[];
  unsigned short* alpha = (unsigned short*)smem_raw;   // [BROWS][QDIM] bf16
  unsigned short* xp    = alpha + BROWS * QDIM;        // [BROWS][SPAD] bf16
  unsigned short* Bl    = xp + BROWS * SPAD;           // [QDIM][SPAD]  bf16
  float* cl = (float*)(Bl + QDIM * SPAD);              // [BROWS] per-step scale
  float* ll = cl + BROWS;                              // [BROWS] running loglik

  const int tid   = threadIdx.x;
  const int lane  = tid & 31;
  const int wave  = tid >> 5;                          // 0..15
  const int m     = blockIdx.x >> 1;
  const int bbase = (blockIdx.x & 1) * BROWS;

  const int n  = lane & 15;                            // WMMA column within tile
  const int hi = lane >> 4;                            // lane half

  // Stage padded emission matrix for this model into LDS (time-invariant).
  const unsigned short* Bpm = Bp + (size_t)m * QDIM * SPAD;
  for (int i = tid; i < QDIM * SPAD; i += 512) Bl[i] = Bpm[i];
  if (tid < BROWS) { cl[tid] = 0.0f; ll[tid] = 0.0f; }
  __syncthreads();

  // Persistent emission B-fragments (K=32) for this wave's two p-tiles.
  const int P0 = wave * 32;
  FragB be0, be1;
  {
    const unsigned short* p0 = Bl + (P0 + n) * SPAD + hi * 16;
    be0.d[0] = *(const i4x*)(p0);
    be0.d[1] = *(const i4x*)(p0 + 8);
    const unsigned short* p1 = Bl + (P0 + 16 + n) * SPAD + hi * 16;
    be1.d[0] = *(const i4x*)(p1);
    be1.d[1] = *(const i4x*)(p1 + 8);
  }

  const float iv0 = initp[m * QDIM + P0 + n];
  const float iv1 = initp[m * QDIM + P0 + 16 + n];

  const unsigned short* Atm  = At + (size_t)m * QDIM * QDIM;
  const unsigned short* atr0 = Atm + (size_t)(P0 + n) * QDIM + hi * 16;
  const unsigned short* atr1 = Atm + (size_t)(P0 + 16 + n) * QDIM + hi * 16;

  // Each thread stages one x element per step: row=wave, s=lane.
  const float* xrow = x + (size_t)(m * BATCH + bbase + wave) * TSTEPS * SDIM;

  for (int t = 0; t < TSTEPS; ++t) {
    // --- stage x_t tile (16x32 bf16) ---
    unsigned short xv = 0;
    if (lane < SDIM) {
      xv = f2bf(xrow[t * SDIM + lane]);
      // WGP-scope prefetch of next step's x row: a full timestep of lead
      // time, sits right before a barrier -> next step's load hits WGP$.
      if (t + 1 < TSTEPS) prefetch_wgp(&xrow[(t + 1) * SDIM + lane]);
    }
    xp[wave * SPAD + lane] = xv;
    __syncthreads();

    // --- emissions via WMMA: em = x_t (16x32) * Be^T (32x16) ---
    FragB xf;
    {
      const unsigned short* p = xp + n * SPAD + hi * 16;
      xf.d[0] = *(const i4x*)(p);
      xf.d[1] = *(const i4x*)(p + 8);
    }
    v8f z = {};
    v8f em0 = __builtin_amdgcn_wmma_f32_16x16x32_bf16(false, xf.v, false, be0.v,
                                                      (short)0, z, false, false);
    v8f em1 = __builtin_amdgcn_wmma_f32_16x16x32_bf16(false, xf.v, false, be1.v,
                                                      (short)0, z, false, false);

    v8f a0, a1;
    if (t == 0) {
      // alpha0 = init ⊙ em0 (init broadcast along the p column of each lane)
      #pragma unroll
      for (int i = 0; i < 8; ++i) { a0[i] = em0[i] * iv0; a1[i] = em1[i] * iv1; }
    } else {
      // --- transition GEMM: alpha (16x512) * At^T-slabs, K=512 ---
      // No prefetch here: it cannot lead the demand stream by more than a
      // few dozen cycles and was observed to break the scheduler's two-slab
      // load pipeline (8 loads in flight across the WMMAs).
      v8f acc0 = {}; v8f acc1 = {};
      #pragma unroll 4
      for (int k0 = 0; k0 < QDIM; k0 += 32) {
        FragB af, bf0, bf1;
        const unsigned short* pa = alpha + n * QDIM + k0 + hi * 16;
        af.d[0] = *(const i4x*)(pa);
        af.d[1] = *(const i4x*)(pa + 8);
        const unsigned short* pb0 = atr0 + k0;
        bf0.d[0] = *(const i4x*)(pb0);
        bf0.d[1] = *(const i4x*)(pb0 + 8);
        const unsigned short* pb1 = atr1 + k0;
        bf1.d[0] = *(const i4x*)(pb1);
        bf1.d[1] = *(const i4x*)(pb1 + 8);
        acc0 = __builtin_amdgcn_wmma_f32_16x16x32_bf16(false, af.v, false, bf0.v,
                                                       (short)0, acc0, false, false);
        acc1 = __builtin_amdgcn_wmma_f32_16x16x32_bf16(false, af.v, false, bf1.v,
                                                       (short)0, acc1, false, false);
      }
      #pragma unroll
      for (int i = 0; i < 8; ++i) { a0[i] = acc0[i] * em0[i]; a1[i] = acc1[i] * em1[i]; }
    }

    // --- per-row scale c[b] = sum_p a[b][p]: butterfly over 16 lanes + LDS add ---
    float rs[8];
    #pragma unroll
    for (int i = 0; i < 8; ++i) {
      float s = a0[i] + a1[i];
      s += __shfl_xor(s, 1, 32);
      s += __shfl_xor(s, 2, 32);
      s += __shfl_xor(s, 4, 32);
      s += __shfl_xor(s, 8, 32);
      rs[i] = s;
    }
    if (n == 0) {
      #pragma unroll
      for (int i = 0; i < 8; ++i) atomicAdd(&cl[i + hi * 8], rs[i]);
    }
    __syncthreads();

    // --- normalize, write next alpha (bf16, D-tile layout -> LDS rows) ---
    #pragma unroll
    for (int i = 0; i < 8; ++i) {
      const int row = i + hi * 8;
      const float rc = 1.0f / cl[row];
      alpha[row * QDIM + P0 + n]      = f2bf(a0[i] * rc);
      alpha[row * QDIM + P0 + 16 + n] = f2bf(a1[i] * rc);
    }
    if (tid < BROWS) ll[tid] += logf(cl[tid]);
    __syncthreads();
    if (tid < BROWS) cl[tid] = 0.0f;   // re-armed before next step's atomics by sync above
  }

  if (tid < BROWS) out[m * BATCH + bbase + tid] = ll[tid];
}

extern "C" void kernel_launch(void* const* d_in, const int* in_sizes, int n_in,
                              void* d_out, int out_size, void* d_ws, size_t ws_size,
                              hipStream_t stream) {
  (void)in_sizes; (void)n_in; (void)out_size; (void)ws_size;
  const float* x    = (const float*)d_in[0];
  const float* A    = (const float*)d_in[1];
  const float* B    = (const float*)d_in[2];
  const float* init = (const float*)d_in[3];

  unsigned short* At = (unsigned short*)d_ws;                       // 2 MB
  unsigned short* Bp = At + (size_t)MDIM * QDIM * QDIM;             // +128 KB

  const int totalConv = MDIM * QDIM * QDIM + MDIM * QDIM * SPAD;
  hmm_convert_kernel<<<(totalConv + 255) / 256, 256, 0, stream>>>(A, B, At, Bp);

  const size_t smem =
      (size_t)(BROWS * QDIM + BROWS * SPAD + QDIM * SPAD) * sizeof(unsigned short)
      + 2 * BROWS * sizeof(float);
  hmm_forward_kernel<<<MDIM * 2, 512, smem, stream>>>(x, init, At, Bp,
                                                      (float*)d_out);
}
